// MultigateMixtureOfExperts_49134425866511
// MI455X (gfx1250) — matrix operand order repaired and checked by reading, not compiled, not run on hardware
//
#include <hip/hip_runtime.h>
#include <hip/hip_bf16.h>

// ---------------------------------------------------------------------------
// MMoE fused kernel for gfx1250 (CDNA5, wave32, WMMA).
//   B=32768, I=512, E=8, X=128, T=2, W=64
// v5: revert to v3 loop structure (best register allocation / instruction
//     count observed), minus the in-loop prefetches that caused the residual
//     8-byte address spills. BM=64, fp16 x-split (x_hi+x_lo) @ fp16 W,
//     ds_add_f32 combine into Sh, towers in f32 VALU.
// ---------------------------------------------------------------------------

typedef __attribute__((ext_vector_type(16))) _Float16 v16h;
typedef __attribute__((ext_vector_type(8)))  float    v8f;

#define B_SZ   32768
#define I_SZ   512
#define E_SZ   8
#define X_SZ   128
#define T_SZ   2
#define W_SZ   64
#define NCOL   1040      // E*X expert cols + T*E gate cols
#define BM     64        // batch rows per block
#define KT     32        // K step (WMMA f16 K)

union FragU { unsigned u[8]; v16h v; };

// ---------------------------------------------------------------------------
// Pre-pass: build transposed fp16 weight matrix Bt[1040][512].
//   n <  1024 : Bt[n][k] = We[e][k][xc]   (e = n>>7, xc = n&127)
//   n >= 1024 : Bt[n][k] = gates[t][k][e] (m = n-1024, t = m>>3, e = m&7)
// ---------------------------------------------------------------------------
__global__ void mmoe_convert_w(const float* __restrict__ We,
                               const float* __restrict__ gates,
                               _Float16* __restrict__ Bt) {
  int gid = blockIdx.x * 256 + threadIdx.x;
  if (gid >= NCOL * I_SZ) return;
  int n = gid / I_SZ;
  int k = gid - n * I_SZ;
  float v;
  if (n < E_SZ * X_SZ) {
    int e = n >> 7, xc = n & 127;
    v = We[(e * I_SZ + k) * X_SZ + xc];
  } else {
    int m = n - E_SZ * X_SZ;
    int t = m >> 3, e = m & 7;
    v = gates[(t * I_SZ + k) * E_SZ + e];
  }
  Bt[gid] = (_Float16)v;
}

// A fragment (16x32 f16) from LDS, row-major [16][KT] at `base`.
__device__ __forceinline__ v16h load_a_frag(const _Float16* base, int lane) {
  int r  = lane & 15;
  int ko = (lane & 16) ? 8 : 0;     // lanes 0-15: K0..7/16..23, 16-31: K8..15/24..31
  const _Float16* p = base + r * KT + ko;
  FragU f;
  *(uint4*)&f.u[0] = *(const uint4*)(p);        // K ko..ko+7
  *(uint4*)&f.u[4] = *(const uint4*)(p + 16);   // K ko+16..ko+23
  return f.v;
}

// B fragment (32x16 f16) from global Bt[NCOL][512] (row n = output col).
__device__ __forceinline__ v16h load_b_frag(const _Float16* __restrict__ Bt,
                                            int ncol0, int k0, int lane) {
  int n  = ncol0 + (lane & 15);
  int ks = k0 + ((lane & 16) ? 8 : 0);          // lanes 0-15: K0..7/16..23, 16-31: K8..15/24..31
  const _Float16* p = Bt + n * I_SZ + ks;
  FragU f;
  *(uint4*)&f.u[0] = *(const uint4*)(p);        // K ks..ks+7
  *(uint4*)&f.u[4] = *(const uint4*)(p + 16);   // K ks+16..ks+23
  return f.v;
}

// ---------------------------------------------------------------------------
// Fused MMoE kernel. Grid: B/BM = 512 blocks. Block: 1024 threads (32 waves).
//   Wave w owns expert n-tiles {2w, 2w+1} (cols w*32 .. w*32+31).
//   Waves 28..31 additionally own gate tile 64, M-tile (w-28).
// ---------------------------------------------------------------------------
__global__ __launch_bounds__(1024)
void mmoe_fused(const float* __restrict__ x,
                const float* __restrict__ be,
                const float* __restrict__ tW1,
                const float* __restrict__ tb1,
                const float* __restrict__ tW2,
                const float* __restrict__ tb2,
                const _Float16* __restrict__ Bt,
                float* __restrict__ out) {
  __shared__ __align__(16) _Float16 As_hi[BM * KT];         // 4 KB
  __shared__ __align__(16) _Float16 As_lo[BM * KT];         // 4 KB
  __shared__ float Gg[BM * 16];                             // 4 KB:  x@gates
  __shared__ float Sh[T_SZ * BM * X_SZ];                    // 64 KB: combined
  __shared__ float Hh[T_SZ * BM * W_SZ];                    // 32 KB: tower hidden

  const int tid     = threadIdx.x;
  const int lane    = tid & 31;
  const int wave    = tid >> 5;
  const int rowbase = blockIdx.x * BM;
  const bool has_gate = (wave >= 28);     // wave-uniform
  const int gate_mt   = wave - 28;

  v8f acc[2][4] = {};     // [n-tile][M-tile], 64 VGPRs/lane
  v8f gacc = {};          // gate tile accumulator (waves 28..31)

  // zero the combine buffer once (used only after the K loop + barriers)
  for (int idx = tid; idx < T_SZ * BM * X_SZ; idx += 1024) Sh[idx] = 0.0f;

  for (int k0 = 0; k0 < I_SZ; k0 += KT) {
    __syncthreads();   // previous iter's LDS fragment reads done
    // ---- stage A tile: f32 -> packed fp16 hi/lo (one k-pair per thread) ----
    {
      int p  = tid;                   // pair index, 0..1023
      int r  = p >> 4;                // row 0..63
      int kp = p & 15;                // k-pair 0..15
      const float2 xv = *(const float2*)&x[(rowbase + r) * I_SZ + k0 + kp * 2];
      _Float16 h0 = (_Float16)xv.x, h1 = (_Float16)xv.y;
      _Float16 l0 = (_Float16)(xv.x - (float)h0);
      _Float16 l1 = (_Float16)(xv.y - (float)h1);
      union { _Float16 h[2]; unsigned u; } ph, pl;
      ph.h[0] = h0; ph.h[1] = h1;
      pl.h[0] = l0; pl.h[1] = l1;
      ((unsigned*)As_hi)[p] = ph.u;
      ((unsigned*)As_lo)[p] = pl.u;
    }
    __syncthreads();

    // ---- expert tiles: acc += (x_hi + x_lo) @ W ----
#pragma unroll
    for (int j = 0; j < 2; ++j) {
      v16h b = load_b_frag(Bt, wave * 32 + j * 16, k0, lane);
#pragma unroll
      for (int mt = 0; mt < 4; ++mt) {
        v16h a_hi = load_a_frag(As_hi + mt * 16 * KT, lane);
        v16h a_lo = load_a_frag(As_lo + mt * 16 * KT, lane);
        acc[j][mt] = __builtin_amdgcn_wmma_f32_16x16x32_f16(
            false, a_hi, false, b, (short)0, acc[j][mt], false, false);
        acc[j][mt] = __builtin_amdgcn_wmma_f32_16x16x32_f16(
            false, a_lo, false, b, (short)0, acc[j][mt], false, false);
      }
    }
    // ---- gate tile (cols 1024..1039), waves 28..31, one M-tile each ----
    if (has_gate) {
      v16h b    = load_b_frag(Bt, 1024, k0, lane);
      v16h a_hi = load_a_frag(As_hi + gate_mt * 16 * KT, lane);
      v16h a_lo = load_a_frag(As_lo + gate_mt * 16 * KT, lane);
      gacc = __builtin_amdgcn_wmma_f32_16x16x32_f16(
          false, a_hi, false, b, (short)0, gacc, false, false);
      gacc = __builtin_amdgcn_wmma_f32_16x16x32_f16(
          false, a_lo, false, b, (short)0, gacc, false, false);
    }
  }

  // ---- store gate tiles: Gg[m][t*8+e] ----
  if (has_gate) {
    int nl   = lane & 15;                      // = t*8 + e
    int mrow = gate_mt * 16 + ((lane & 16) ? 8 : 0);
#pragma unroll
    for (int v = 0; v < 8; ++v)
      Gg[(mrow + v) * 16 + nl] = gacc[v];
  }
  __syncthreads();

  // ---- combine: Sh[t][m][xc] += Gg[m][t*8+e] * relu(acc + be[n]) ----
#pragma unroll
  for (int j = 0; j < 2; ++j) {
    int n  = wave * 32 + j * 16 + (lane & 15);
    int e  = n >> 7;
    int xc = n & 127;
    float bias = be[n];                        // be is [E*X] = [1024] flat
#pragma unroll
    for (int mt = 0; mt < 4; ++mt) {
      int mrow = mt * 16 + ((lane & 16) ? 8 : 0);
#pragma unroll
      for (int v = 0; v < 8; ++v) {
        int m = mrow + v;
        float val = fmaxf(acc[j][mt][v] + bias, 0.0f);
        atomicAdd(&Sh[m * X_SZ + xc],             Gg[m * 16 + e]     * val);
        atomicAdd(&Sh[BM * X_SZ + m * X_SZ + xc], Gg[m * 16 + 8 + e] * val);
      }
    }
  }
  __syncthreads();

  // ---- tower layer 1: Hh[t][b][w] = relu(Sh[t][b][:] . tW1[t][:,w] + tb1) ----
  for (int idx = tid; idx < T_SZ * BM * W_SZ; idx += 1024) {
    int wc = idx & 63;
    int bb = (idx >> 6) & (BM - 1);
    int t  = idx >> 12;
    float s = tb1[t * W_SZ + wc];
    const float* shp = &Sh[(t * BM + bb) * X_SZ];
    const float* w1  = &tW1[t * X_SZ * W_SZ + wc];
#pragma unroll 8
    for (int xc = 0; xc < X_SZ; ++xc)
      s += shp[xc] * w1[xc * W_SZ];
    Hh[idx] = fmaxf(s, 0.0f);   // layout [t][b][w]
  }
  __syncthreads();

  // ---- tower layer 2: out[t][b] = Hh[t][b][:] . tW2[t][:] + tb2[t] ----
  if (tid < T_SZ * BM) {
    int bb = tid & (BM - 1);
    int t  = tid >> 6;
    float s = tb2[t];
    const float* hp = &Hh[(t * BM + bb) * W_SZ];
    const float* w2 = &tW2[t * W_SZ];
#pragma unroll 8
    for (int wc = 0; wc < W_SZ; ++wc)
      s += hp[wc] * w2[wc];
    out[t * B_SZ + rowbase + bb] = s;
  }
}

// ---------------------------------------------------------------------------
extern "C" void kernel_launch(void* const* d_in, const int* in_sizes, int n_in,
                              void* d_out, int out_size, void* d_ws, size_t ws_size,
                              hipStream_t stream) {
  const float* x     = (const float*)d_in[0];
  const float* We    = (const float*)d_in[1];
  const float* be    = (const float*)d_in[2];
  const float* gates = (const float*)d_in[3];
  const float* tW1   = (const float*)d_in[4];
  const float* tb1   = (const float*)d_in[5];
  const float* tW2   = (const float*)d_in[6];
  const float* tb2   = (const float*)d_in[7];
  float* out = (float*)d_out;

  _Float16* Bt = (_Float16*)d_ws;

  // Pre-pass: transpose + fp16-convert the (tiny, L2-resident) weights.
  {
    int total = NCOL * I_SZ;
    mmoe_convert_w<<<(total + 255) / 256, 256, 0, stream>>>(We, gates, Bt);
  }
  // Fused GEMM + gating + towers.
  mmoe_fused<<<B_SZ / BM, 1024, 0, stream>>>(x, be, tW1, tb1, tW2, tb2, Bt, out);
}
